// Seq2Seq_69071664054911
// MI455X (gfx1250) — matrix operand order repaired and checked by reading, not compiled, not run on hardware
//
#include <hip/hip_runtime.h>

#define DEV __device__ __forceinline__

typedef __attribute__((ext_vector_type(16))) __bf16        v16bf;
typedef __attribute__((ext_vector_type(8)))  float         v8f;
typedef __attribute__((ext_vector_type(4)))  unsigned int  u32x4;

union FragU { v16bf v; u32x4 q[2]; };

constexpr int V_ = 32000, E_ = 512, H_ = 1024, B_ = 32, S_ = 64, T_ = 48;
constexpr int H3 = 3 * H_;      // 3072
constexpr int KX = E_ + H_;     // 1536 (decoder GRU input width)
constexpr int K2 = 2 * H_;      // 2048 (logits input width)
constexpr int TO = T_ - 1;      // 47 decoder steps

// ---------------- helpers ----------------

DEV unsigned short f32_to_bf16(float f) {
  unsigned int u = __float_as_uint(f);
  u += 0x7fffu + ((u >> 16) & 1u);          // round-to-nearest-even
  return (unsigned short)(u >> 16);
}
DEV float sigmoidf_(float x) { return 1.0f / (1.0f + __expf(-x)); }

// A fragment: 16x32 bf16 tile, rows r0..r0+15, K = k0..k0+31, row-major, ld elems.
// Contiguous K order (same as B): lane<16 -> row M=lane, K=0..15; lane>=16 -> K=16..31.
DEV v16bf load_frag_a(const unsigned short* p, int r0, int k0, int ld, int lane) {
  int m = lane & 15, half = lane >> 4;
  const u32x4* row = (const u32x4*)(p + (size_t)(r0 + m) * ld + k0);
  FragU f; f.q[0] = row[2 * half]; f.q[1] = row[2 * half + 1];
  return f.v;
}
// B fragment: K x N = 32x16, sourced from W[n][k] row-major (rows = output cols).
// Contiguous K order (same as A): lane<16 -> col N=lane, K=0..15; lane>=16 -> col N=lane-16, K=16..31.
DEV v16bf load_frag_b(const unsigned short* p, int n0, int k0, int ld, int lane) {
  int n = lane & 15, half = lane >> 4;
  const u32x4* row = (const u32x4*)(p + (size_t)(n0 + n) * ld + k0 + (half << 4));
  FragU f; f.q[0] = row[0]; f.q[1] = row[1];
  return f.v;
}
DEV v8f wmma_bf16(v16bf a, v16bf b, v8f c) {
  return __builtin_amdgcn_wmma_f32_16x16x32_bf16(false, a, false, b, (short)0, c,
                                                 false, false);
}

// ---------------- setup kernels ----------------

__global__ void cvt_kernel(const float* __restrict__ s, unsigned short* __restrict__ d,
                           size_t n) {
  size_t i = (size_t)blockIdx.x * blockDim.x + threadIdx.x;
  size_t stride = (size_t)gridDim.x * blockDim.x;
  for (; i < n; i += stride) d[i] = f32_to_bf16(s[i]);
}

__global__ void embed_src_kernel(const int* __restrict__ src,
                                 const float* __restrict__ E_enc,
                                 unsigned short* __restrict__ emb_bf) {
  int i = blockIdx.x * blockDim.x + threadIdx.x;   // exactly B*S*E threads
  int row = i / E_, k = i - row * E_;
  int tok = src[row];
  emb_bf[i] = f32_to_bf16(E_enc[(size_t)tok * E_ + k]);
}

__global__ void zero_h_kernel(float* __restrict__ hf, unsigned short* __restrict__ hb) {
  int i = blockIdx.x * blockDim.x + threadIdx.x;
  if (i < B_ * H_) { hf[i] = 0.f; hb[i] = 0; }
}

// -------- 2x2 register-blocked WMMA GEMM with bias (C = A*W^T + bias) --------
// one wave per 32x32 output tile: 4 WMMAs per k-step from 8 b128 loads (2 loads/WMMA)
__global__ void gemm_bias32_kernel(const unsigned short* __restrict__ A, int lda,
                                   const unsigned short* __restrict__ Bw, int ldb,
                                   const float* __restrict__ bias,
                                   float* __restrict__ C, int ldc,
                                   int MT2, int NT2, int K) {
  int gid = blockIdx.x * blockDim.x + threadIdx.x;
  int wave = gid >> 5, lane = threadIdx.x & 31;
  if (wave >= MT2 * NT2) return;
  int tn = wave % NT2, tm = wave / NT2;
  int r0 = tm * 32, c0 = tn * 32;
  v8f acc00, acc01, acc10, acc11;
  {
    float b0 = bias[c0 + (lane & 15)];
    float b1 = bias[c0 + 16 + (lane & 15)];
    for (int v = 0; v < 8; ++v) { acc00[v] = b0; acc01[v] = b1; acc10[v] = b0; acc11[v] = b1; }
  }
  for (int k = 0; k < K; k += 32) {
    v16bf a0 = load_frag_a(A, r0,      k, lda, lane);
    v16bf a1 = load_frag_a(A, r0 + 16, k, lda, lane);
    v16bf b0 = load_frag_b(Bw, c0,      k, ldb, lane);
    v16bf b1 = load_frag_b(Bw, c0 + 16, k, ldb, lane);
    acc00 = wmma_bf16(a0, b0, acc00);
    acc01 = wmma_bf16(a0, b1, acc01);
    acc10 = wmma_bf16(a1, b0, acc10);
    acc11 = wmma_bf16(a1, b1, acc11);
  }
  int n = lane & 15, mb = (lane >> 4) * 8;
  for (int v = 0; v < 8; ++v) {
    size_t rA = (size_t)(r0 + mb + v) * ldc;
    size_t rB = (size_t)(r0 + 16 + mb + v) * ldc;
    C[rA + c0 + n]      = acc00[v];
    C[rA + c0 + 16 + n] = acc01[v];
    C[rB + c0 + n]      = acc10[v];
    C[rB + c0 + 16 + n] = acc11[v];
  }
}

// ---------------- encoder recurrent step ----------------
// 128 waves: wave = (tm in 0..1, tn in 0..63), tile of h_new[32,1024]
__global__ void enc_step_kernel(const unsigned short* __restrict__ h_bf_in,
                                const float* __restrict__ h_f32_in,
                                unsigned short* __restrict__ h_bf_out,
                                float* __restrict__ h_f32_out,
                                const unsigned short* __restrict__ Whh_bf,
                                const float* __restrict__ gi_all,
                                const float* __restrict__ bhh,
                                const int* __restrict__ src_lens,
                                float* __restrict__ enc_out, int t) {
  int gid = blockIdx.x * blockDim.x + threadIdx.x;
  int wave = gid >> 5, lane = threadIdx.x & 31;
  if (wave >= 2 * (H_ / 16)) return;
  int tm = wave & 1, tn = wave >> 1;
  int r0 = tm * 16, c0 = tn * 16;

  v8f gh[3];
  for (int g = 0; g < 3; ++g) {
    int col0 = g * H_ + c0;
    float bh = bhh[col0 + (lane & 15)];
    v8f a; for (int v = 0; v < 8; ++v) a[v] = bh;
    for (int k = 0; k < H_; k += 32)
      a = wmma_bf16(load_frag_a(h_bf_in, r0, k, H_, lane),
                    load_frag_b(Whh_bf, col0, k, H_, lane), a);
    gh[g] = a;
  }
  int n = lane & 15, mb = (lane >> 4) * 8;
  for (int v = 0; v < 8; ++v) {
    int b = r0 + mb + v;
    const float* gir = gi_all + (size_t)(b * S_ + t) * H3 + c0 + n;  // has bih
    float hprev = h_f32_in[b * H_ + c0 + n];
    float r  = sigmoidf_(gir[0]      + gh[0][v]);
    float z  = sigmoidf_(gir[H_]     + gh[1][v]);
    float nn = tanhf   (gir[2 * H_]  + r * gh[2][v]);
    float hnew = (1.f - z) * nn + z * hprev;
    bool valid = t < src_lens[b];
    float hk = valid ? hnew : hprev;                       // freeze past length
    h_f32_out[b * H_ + c0 + n] = hk;
    h_bf_out [b * H_ + c0 + n] = f32_to_bf16(hk);
    enc_out[((size_t)b * S_ + t) * H_ + c0 + n] = valid ? hnew : 0.f;
  }
}

// ---------------- decoder attention + input build (per step) ----------------
// grid = B blocks of 256 threads
__global__ void attn_kernel(const float* __restrict__ enc_out,
                            const float* __restrict__ h_f32_in,
                            const int* __restrict__ src,
                            const int* __restrict__ tgt,
                            const float* __restrict__ E_dec,
                            unsigned short* __restrict__ x_bf,    // [B, KX]
                            unsigned short* __restrict__ x2_bf,   // [B, K2]
                            int t) {
  __shared__ float part[S_ * 4];
  __shared__ float wgt[S_];
  __shared__ float red;
  int b = blockIdx.x, tid = threadIdx.x;
  int s = tid >> 2, l4 = tid & 3;
  const float* eo = enc_out + ((size_t)b * S_ + s) * H_;
  const float* hb = h_f32_in + b * H_;
  float p = 0.f;
  for (int k = l4; k < H_; k += 4) p += eo[k] * hb[k];
  part[s * 4 + l4] = p;
  __syncthreads();
  if (tid < S_) {
    float sc = part[tid * 4] + part[tid * 4 + 1] + part[tid * 4 + 2] + part[tid * 4 + 3];
    wgt[tid] = (src[b * S_ + tid] != 0) ? sc : -1e9f;      // attention mask
  }
  __syncthreads();
  if (tid == 0) {
    float mx = -1e30f;
    for (int i = 0; i < S_; ++i) mx = fmaxf(mx, wgt[i]);
    float sum = 0.f;
    for (int i = 0; i < S_; ++i) { float e = __expf(wgt[i] - mx); wgt[i] = e; sum += e; }
    red = 1.f / sum;
  }
  __syncthreads();
  float inv = red;
  for (int j = tid; j < H_; j += 256) {
    float c = 0.f;
    for (int s2 = 0; s2 < S_; ++s2)
      c += wgt[s2] * enc_out[((size_t)b * S_ + s2) * H_ + j];
    unsigned short cb = f32_to_bf16(c * inv);
    x_bf [b * KX + E_ + j]  = cb;       // GRU input: [emb | ctx]
    x2_bf[b * K2 + H_ + j]  = cb;       // logits input: [h_new | ctx]
  }
  int tok = tgt[b * T_ + t];
  for (int j = tid; j < E_; j += 256)
    x_bf[b * KX + j] = f32_to_bf16(E_dec[(size_t)tok * E_ + j]);
}

// ---------------- decoder GRU step ----------------
// 128 waves, same tiling as enc_step; computes gi (K=1536) and gh (K=1024) per gate
__global__ void dec_gru_kernel(const unsigned short* __restrict__ x_bf,
                               const unsigned short* __restrict__ h_bf_in,
                               const float* __restrict__ h_f32_in,
                               unsigned short* __restrict__ h_bf_out,
                               float* __restrict__ h_f32_out,
                               unsigned short* __restrict__ x2_bf,
                               const unsigned short* __restrict__ Wih_bf,
                               const unsigned short* __restrict__ Whh_bf,
                               const float* __restrict__ bih,
                               const float* __restrict__ bhh) {
  int gid = blockIdx.x * blockDim.x + threadIdx.x;
  int wave = gid >> 5, lane = threadIdx.x & 31;
  if (wave >= 2 * (H_ / 16)) return;
  int tm = wave & 1, tn = wave >> 1;
  int r0 = tm * 16, c0 = tn * 16;

  v8f gi[3], gh[3];
  for (int g = 0; g < 3; ++g) {
    int col0 = g * H_ + c0;
    float bi = bih[col0 + (lane & 15)];
    float bh = bhh[col0 + (lane & 15)];
    v8f ai, ah;
    for (int v = 0; v < 8; ++v) { ai[v] = bi; ah[v] = bh; }
    for (int k = 0; k < KX; k += 32)
      ai = wmma_bf16(load_frag_a(x_bf, r0, k, KX, lane),
                     load_frag_b(Wih_bf, col0, k, KX, lane), ai);
    for (int k = 0; k < H_; k += 32)
      ah = wmma_bf16(load_frag_a(h_bf_in, r0, k, H_, lane),
                     load_frag_b(Whh_bf, col0, k, H_, lane), ah);
    gi[g] = ai; gh[g] = ah;
  }
  int n = lane & 15, mb = (lane >> 4) * 8;
  for (int v = 0; v < 8; ++v) {
    int b = r0 + mb + v;
    float hprev = h_f32_in[b * H_ + c0 + n];
    float r  = sigmoidf_(gi[0][v] + gh[0][v]);
    float z  = sigmoidf_(gi[1][v] + gh[1][v]);
    float nn = tanhf   (gi[2][v] + r * gh[2][v]);
    float hnew = (1.f - z) * nn + z * hprev;
    h_f32_out[b * H_ + c0 + n] = hnew;
    unsigned short h16 = f32_to_bf16(hnew);
    h_bf_out[b * H_ + c0 + n] = h16;
    x2_bf[b * K2 + c0 + n] = h16;        // logits input columns 0..H-1
  }
}

// ---------------- logits: [32,2048] x [2048,32000] per step ----------------
// 2x2 blocked: one wave per 32x32 tile (all of M x two N tiles).
// 1000 waves/step, 256 WMMAs each, K=2048; out_W bf16 is L2-resident (131 MB < 192 MB)
__global__ void logits_kernel(const unsigned short* __restrict__ x2_bf,
                              const unsigned short* __restrict__ outW_bf,
                              const float* __restrict__ out_b,
                              float* __restrict__ out, int t) {
  int gid = blockIdx.x * blockDim.x + threadIdx.x;
  int wave = gid >> 5, lane = threadIdx.x & 31;
  if (wave >= V_ / 32) return;
  int c0 = wave * 32;
  v8f acc00, acc01, acc10, acc11;
  {
    float b0 = out_b[c0 + (lane & 15)];
    float b1 = out_b[c0 + 16 + (lane & 15)];
    for (int v = 0; v < 8; ++v) { acc00[v] = b0; acc01[v] = b1; acc10[v] = b0; acc11[v] = b1; }
  }
  for (int k = 0; k < K2; k += 32) {
    v16bf a0 = load_frag_a(x2_bf, 0,  k, K2, lane);
    v16bf a1 = load_frag_a(x2_bf, 16, k, K2, lane);
    v16bf b0 = load_frag_b(outW_bf, c0,      k, K2, lane);
    v16bf b1 = load_frag_b(outW_bf, c0 + 16, k, K2, lane);
    acc00 = wmma_bf16(a0, b0, acc00);
    acc01 = wmma_bf16(a0, b1, acc01);
    acc10 = wmma_bf16(a1, b0, acc10);
    acc11 = wmma_bf16(a1, b1, acc11);
  }
  int n = lane & 15, mb = (lane >> 4) * 8;
  for (int v = 0; v < 8; ++v) {
    size_t rA = ((size_t)(mb + v) * TO + t) * V_;        // batch rows 0..15
    size_t rB = ((size_t)(16 + mb + v) * TO + t) * V_;   // batch rows 16..31
    out[rA + c0 + n]      = acc00[v];
    out[rA + c0 + 16 + n] = acc01[v];
    out[rB + c0 + n]      = acc10[v];
    out[rB + c0 + 16 + n] = acc11[v];
  }
}

// ---------------- host ----------------

extern "C" void kernel_launch(void* const* d_in, const int* in_sizes, int n_in,
                              void* d_out, int out_size, void* d_ws, size_t ws_size,
                              hipStream_t stream) {
  (void)in_sizes; (void)n_in; (void)out_size; (void)ws_size;
  const int*   src      = (const int*)d_in[0];
  const int*   src_lens = (const int*)d_in[1];
  const int*   tgt      = (const int*)d_in[2];
  const float* E_enc    = (const float*)d_in[3];
  const float* enc_Wih  = (const float*)d_in[4];
  const float* enc_Whh  = (const float*)d_in[5];
  const float* enc_bih  = (const float*)d_in[6];
  const float* enc_bhh  = (const float*)d_in[7];
  const float* E_dec    = (const float*)d_in[8];
  const float* dec_Wih  = (const float*)d_in[9];
  const float* dec_Whh  = (const float*)d_in[10];
  const float* dec_bih  = (const float*)d_in[11];
  const float* dec_bhh  = (const float*)d_in[12];
  const float* out_W    = (const float*)d_in[13];
  const float* out_b    = (const float*)d_in[14];
  float* out = (float*)d_out;

  char* base = (char*)d_ws;
  size_t off = 0;
  auto alloc = [&](size_t bytes) -> char* {
    char* p = base + off;
    off += (bytes + 255) & ~(size_t)255;
    return p;
  };
  unsigned short* outW_bf = (unsigned short*)alloc((size_t)V_ * K2 * 2);  // 131 MB, fits L2
  unsigned short* eWih_bf = (unsigned short*)alloc((size_t)H3 * E_ * 2);
  unsigned short* eWhh_bf = (unsigned short*)alloc((size_t)H3 * H_ * 2);
  unsigned short* dWih_bf = (unsigned short*)alloc((size_t)H3 * KX * 2);
  unsigned short* dWhh_bf = (unsigned short*)alloc((size_t)H3 * H_ * 2);
  unsigned short* emb_bf  = (unsigned short*)alloc((size_t)B_ * S_ * E_ * 2);
  float*          gi_all  = (float*)alloc((size_t)B_ * S_ * H3 * 4);
  float*          enc_out = (float*)alloc((size_t)B_ * S_ * H_ * 4);
  float*          hf      = (float*)alloc((size_t)2 * B_ * H_ * 4);  // ping-pong
  unsigned short* hb      = (unsigned short*)alloc((size_t)2 * B_ * H_ * 2);
  unsigned short* x_bf    = (unsigned short*)alloc((size_t)B_ * KX * 2);
  unsigned short* x2_bf   = (unsigned short*)alloc((size_t)B_ * K2 * 2);

  // --- one-time (per call) weight conversion + embedding gather ---
  cvt_kernel<<<4096, 256, 0, stream>>>(out_W,   outW_bf, (size_t)V_ * K2);
  cvt_kernel<<<1024, 256, 0, stream>>>(enc_Wih, eWih_bf, (size_t)H3 * E_);
  cvt_kernel<<<1024, 256, 0, stream>>>(enc_Whh, eWhh_bf, (size_t)H3 * H_);
  cvt_kernel<<<1024, 256, 0, stream>>>(dec_Wih, dWih_bf, (size_t)H3 * KX);
  cvt_kernel<<<1024, 256, 0, stream>>>(dec_Whh, dWhh_bf, (size_t)H3 * H_);
  embed_src_kernel<<<(B_ * S_ * E_) / 256, 256, 0, stream>>>(src, E_enc, emb_bf);
  zero_h_kernel<<<(B_ * H_) / 256, 256, 0, stream>>>(hf, hb);

  // --- all encoder input projections in one GEMM: [B*S,E] x [E,3H] + bih ---
  {
    int MT2 = (B_ * S_) / 32, NT2 = H3 / 32;               // 64 x 96 tiles of 32x32
    gemm_bias32_kernel<<<(MT2 * NT2 + 7) / 8, 256, 0, stream>>>(
        emb_bf, E_, eWih_bf, E_, enc_bih, gi_all, H3, MT2, NT2, E_);
  }

  // --- encoder recurrence (64 sequential launches, ping-pong h) ---
  for (int t = 0; t < S_; ++t) {
    int p0 = t & 1, p1 = p0 ^ 1;
    enc_step_kernel<<<16, 256, 0, stream>>>(
        hb + p0 * B_ * H_, hf + p0 * B_ * H_,
        hb + p1 * B_ * H_, hf + p1 * B_ * H_,
        eWhh_bf, gi_all, enc_bhh, src_lens, enc_out, t);
  }

  // --- decoder (47 sequential steps); h parity continues: final enc h is in buf 0 ---
  for (int t = 0; t < TO; ++t) {
    int p0 = t & 1, p1 = p0 ^ 1;
    attn_kernel<<<B_, 256, 0, stream>>>(enc_out, hf + p0 * B_ * H_, src, tgt,
                                        E_dec, x_bf, x2_bf, t);
    dec_gru_kernel<<<16, 256, 0, stream>>>(
        x_bf, hb + p0 * B_ * H_, hf + p0 * B_ * H_,
        hb + p1 * B_ * H_, hf + p1 * B_ * H_,
        x2_bf, dWih_bf, dWhh_bf, dec_bih, dec_bhh);
    logits_kernel<<<125, 256, 0, stream>>>(x2_bf, outW_bf, out_b, out, t);
  }
}
